// BM3D2D_13116830122308
// MI455X (gfx1250) — matrix operand order repaired and checked by reading, not compiled, not run on hardware
//
#include <hip/hip_runtime.h>
#include <hip/hip_bf16.h>

typedef __attribute__((ext_vector_type(16))) _Float16 v16h;
typedef __attribute__((ext_vector_type(8)))  float    v8f;

#define NIMG 8
#define HW   65536      // 256*256
#define Nn   63         // sites per dim
#define NS2  3969       // 63*63
#define NSITES 31752    // 8*3969
#define THR  0.135f     // LAM*SIGMA
#define SIG2 0.0025f    // SIGMA^2
#define PI_F 3.14159265358979f

// workspace layout (in floats)
#define OFF_T    0u
#define OFF_IDX  2032128u            // 8*3969*64
#define OFF_NUM  2540160u            // + 8*3969*16
#define OFF_CNT  3064448u            // + 8*65536

__device__ __forceinline__ float dct8v(int k, int i) {
    float v = cosf(PI_F * (2.0f * i + 1.0f) * (float)k / 16.0f) * 0.5f; // sqrt(2/8)=0.5
    if (k == 0) v *= 0.70710678118f;
    return v;
}
__device__ __forceinline__ float dct16v(int k, int i) {
    float v = cosf(PI_F * (2.0f * i + 1.0f) * (float)k / 32.0f) * 0.35355339059f; // sqrt(2/16)
    if (k == 0) v *= 0.70710678118f;
    return v;
}

// ---------------- K0: zero num/cnt ----------------
__global__ void k0_zero(float* __restrict__ p, int n) {
    int t = blockIdx.x * blockDim.x + threadIdx.x;
    if (t < n) p[t] = 0.0f;
}

// ---------------- K1: 8x8 DCT of every patch ----------------
__global__ __launch_bounds__(256) void k1_dct(const float* __restrict__ x,
                                              float* __restrict__ T) {
    __shared__ float D8l[64];
    int tid = threadIdx.x;
    if (tid < 64) D8l[tid] = dct8v(tid >> 3, tid & 7);
    __syncthreads();

    int t = blockIdx.x * 256 + tid;
    if (t >= NSITES) return;
    int img  = t / NS2;
    int sidx = t - img * NS2;
    int i = sidx / Nn, j = sidx - i * Nn;
    int y0 = 4 * i, x0 = 4 * j;
    const float* xim = x + (size_t)img * HW;

    float p[8][8];
#pragma unroll
    for (int b = 0; b < 8; ++b)
#pragma unroll
        for (int c = 0; c < 8; ++c) {
            float v = xim[(y0 + b) * 256 + x0 + c];
            p[b][c] = fminf(fmaxf(v, 0.0f), 1.0f);
        }
    float tmp[8][8];
#pragma unroll
    for (int a = 0; a < 8; ++a)
#pragma unroll
        for (int c = 0; c < 8; ++c) {
            float s = 0.0f;
#pragma unroll
            for (int b = 0; b < 8; ++b) s += D8l[a * 8 + b] * p[b][c];
            tmp[a][c] = s;
        }
    float* To = T + ((size_t)img * NS2 + sidx) * 64;
#pragma unroll
    for (int a = 0; a < 8; ++a)
#pragma unroll
        for (int d = 0; d < 8; ++d) {
            float s = 0.0f;
#pragma unroll
            for (int c = 0; c < 8; ++c) s += tmp[a][c] * D8l[d * 8 + c]; // T = tmp * D^T
            To[a * 8 + d] = s;
        }
}

// ---------------- K2: 81 distances + stable top-16 ----------------
__global__ __launch_bounds__(128) void k2_match(const float* __restrict__ T,
                                                unsigned* __restrict__ IDX) {
    __shared__ float Tc[64];
    __shared__ float dist[81];
    int site = blockIdx.x;
    int img  = site / NS2;
    int sidx = site - img * NS2;
    int i = sidx / Nn, j = sidx - i * Nn;
    int tid = threadIdx.x;

    const float* Tim = T + (size_t)img * NS2 * 64;
    if (tid < 64) Tc[tid] = Tim[(size_t)sidx * 64 + tid];
    __syncthreads();

    if (tid < 81) {
        int dy = tid / 9 - 4, dx = tid % 9 - 4;
        int cy = min(max(i + dy, 0), Nn - 1);
        int cx = min(max(j + dx, 0), Nn - 1);
        const float* Tcand = Tim + (size_t)(cy * Nn + cx) * 64;
        __builtin_prefetch(Tcand, 0, 1);
        float ds = 0.0f;
#pragma unroll 8
        for (int e = 0; e < 64; ++e) {
            float df = Tcand[e] - Tc[e];
            ds += df * df;
        }
        dist[tid] = ds;
    }
    __syncthreads();

    if (tid == 0) {
        unsigned* out = IDX + (size_t)site * 16;
        for (int k = 0; k < 16; ++k) {
            float best = 3.0e38f;
            int bc = 0;
            for (int c = 0; c < 81; ++c)
                if (dist[c] < best) { best = dist[c]; bc = c; } // stable: lowest index wins ties
            dist[bc] = 3.4e38f;
            int dy = bc / 9 - 4, dx = bc % 9 - 4;
            int cy = min(max(i + dy, 0), Nn - 1);
            int cx = min(max(j + dx, 0), Nn - 1);
            out[k] = ((unsigned)cy << 8) | (unsigned)cx;
        }
    }
}

// ---------------- K3: WMMA group transform + threshold + inverse + scatter ----------------
// one wave (32 lanes) per site; 4 waves per block; NSITES % 4 == 0 so EXEC is full everywhere
__global__ __launch_bounds__(128) void k3_group(const float* __restrict__ T,
                                                const unsigned* __restrict__ IDX,
                                                float* __restrict__ NUM,
                                                float* __restrict__ CNT) {
    __shared__ float smem[320 + 4 * 2048];
    float* D8l = smem;
    float* DKl = smem + 64;
    int tid = threadIdx.x;
    for (int idx = tid; idx < 64; idx += 128) D8l[idx] = dct8v(idx >> 3, idx & 7);
    for (int idx = tid; idx < 256; idx += 128) DKl[idx] = dct16v(idx >> 4, idx & 15);
    __syncthreads();

    int wave = tid >> 5, lane = tid & 31;
    float* gt  = smem + 320 + wave * 2048;  // Gt staging: 16 x 64 f32
    float* den = gt + 1024;                 // den_g staging: 16 x 64 f32

    int site = blockIdx.x * 4 + wave;       // exact grid: site < NSITES always
    int img  = site / NS2;

    const float*    Tim  = T + (size_t)img * NS2 * 64;
    const unsigned* idxs = IDX + (size_t)site * 16;

    int myv[16], mxv[16];
#pragma unroll
    for (int m = 0; m < 16; ++m) {
        unsigned pk = idxs[m];
        myv[m] = (int)(pk >> 8);
        mxv[m] = (int)(pk & 255u);
    }

    // A fragment: DK (16x16) zero-padded to 16x32 in K, contiguous k order (same in B), not AMD's table order:
// lanes 0-15: row=lane, elements 0-15 hold K=0..15 ; lanes 16-31: row=lane-16, K=16..31 padding (zero)
    int  row = lane & 15;
    int  kb  = (lane < 16) ? 0 : 8;
    bool lo  = (lane < 16);
    int  col = lane & 15;
    v16h adk;
#pragma unroll
    for (int e = 0; e < 16; ++e) {
        float v = lo ? DKl[row * 16 + e] : 0.0f;
        adk[e] = (_Float16)v;
    }

    // G3 = DK * group  (group: member k=0..15 rows, 64 coef cols, split into 4 N-tiles)
    v8f acc[4];
#pragma unroll
    for (int t = 0; t < 4; ++t) {
        v16h b;
#pragma unroll
        for (int e = 0; e < 16; ++e) {
            const float* Tm = Tim + (size_t)(myv[e] * Nn + mxv[e]) * 64;
            float g = Tm[t * 16 + col];
            b[e] = lo ? (_Float16)g : (_Float16)0.0f; // K=16..31 is padding -> 0
        }
        v8f c = {};
        acc[t] = __builtin_amdgcn_wmma_f32_16x16x32_f16(false, adk, false, b,
                                                        (short)0, c, false, false);
    }

    // hard threshold |G3| > lam*sigma ; count nnz across whole 16x64 group
    int nnz = 0;
#pragma unroll
    for (int t = 0; t < 4; ++t)
#pragma unroll
        for (int r = 0; r < 8; ++r) {
            float v = acc[t][r];
            bool m = fabsf(v) > THR;
            nnz += m ? 1 : 0;
            acc[t][r] = m ? v : 0.0f;
        }
#pragma unroll
    for (int off = 16; off > 0; off >>= 1) nnz += __shfl_xor(nnz, off, 32);
    float wgt = 1.0f / (SIG2 * fmaxf((float)nnz, 1.0f));

    // stage Gt to LDS (C/D layout -> row-major 16x64) for re-layout as B fragments
    int rb = lo ? 0 : 8;
#pragma unroll
    for (int t = 0; t < 4; ++t)
#pragma unroll
        for (int r = 0; r < 8; ++r)
            gt[(rb + r) * 64 + t * 16 + col] = acc[t][r];
    __syncthreads();

    // den_g = DK * Gt  (reference applies DK again, per its einsum)
    v8f dacc[4];
#pragma unroll
    for (int t = 0; t < 4; ++t) {
        v16h b;
#pragma unroll
        for (int e = 0; e < 16; ++e) {
            float g = gt[e * 64 + t * 16 + col];
            b[e] = lo ? (_Float16)g : (_Float16)0.0f;
        }
        v8f c = {};
        dacc[t] = __builtin_amdgcn_wmma_f32_16x16x32_f16(false, adk, false, b,
                                                         (short)0, c, false, false);
    }
#pragma unroll
    for (int t = 0; t < 4; ++t)
#pragma unroll
        for (int r = 0; r < 8; ++r)
            den[(rb + r) * 64 + t * 16 + col] = dacc[t][r];
    __syncthreads();

    // inverse 2D DCT per member patch + weighted atomic scatter
    int m  = lane >> 1;
    int a0 = (lane & 1) * 4;
    unsigned pk = idxs[m];
    int my = (int)(pk >> 8), mx = (int)(pk & 255u);
    const float* dm = den + m * 64;

    for (int a = a0; a < a0 + 4; ++a) {
        float tmp[8];
#pragma unroll
        for (int c = 0; c < 8; ++c) {
            float s = 0.0f;
#pragma unroll
            for (int b = 0; b < 8; ++b) s += D8l[b * 8 + a] * dm[b * 8 + c]; // D^T on left
            tmp[c] = s;
        }
        int base = img * HW + (4 * my + a) * 256 + 4 * mx;
#pragma unroll
        for (int d = 0; d < 8; ++d) {
            float s = 0.0f;
#pragma unroll
            for (int c = 0; c < 8; ++c) s += tmp[c] * D8l[c * 8 + d]; // D on right
            atomicAdd(&NUM[base + d], wgt * s);
            atomicAdd(&CNT[base + d], wgt);
        }
    }
}

// ---------------- K4: finalize ----------------
__global__ void k4_final(const float* __restrict__ NUM, const float* __restrict__ CNT,
                         float* __restrict__ out, int n) {
    int t = blockIdx.x * blockDim.x + threadIdx.x;
    if (t < n) out[t] = NUM[t] / fmaxf(CNT[t], 1e-8f);
}

extern "C" void kernel_launch(void* const* d_in, const int* in_sizes, int n_in,
                              void* d_out, int out_size, void* d_ws, size_t ws_size,
                              hipStream_t stream) {
    const float* x = (const float*)d_in[0];
    float* ws  = (float*)d_ws;
    float* T   = ws + OFF_T;
    unsigned* IDX = (unsigned*)(ws + OFF_IDX);
    float* NUM = ws + OFF_NUM;
    float* CNT = ws + OFF_CNT;
    float* out = (float*)d_out;

    const int nacc = 2 * NIMG * HW; // num+cnt contiguous
    k0_zero<<<(nacc + 255) / 256, 256, 0, stream>>>(NUM, nacc);
    k1_dct<<<(NSITES + 255) / 256, 256, 0, stream>>>(x, T);
    k2_match<<<NSITES, 128, 0, stream>>>(T, IDX);
    k3_group<<<NSITES / 4, 128, 0, stream>>>(T, IDX, NUM, CNT);
    k4_final<<<(NIMG * HW + 255) / 256, 256, 0, stream>>>(NUM, CNT, out, NIMG * HW);
}
